// ChannelAttentionModel_8521215115893
// MI455X (gfx1250) — compile-verified, matrix operand order checked by reading, not run on hardware
//
#include <hip/hip_runtime.h>
#include <hip/hip_bf16.h>
#include <math.h>

// ---------------------------------------------------------------------------
// ChannelAttention for MI455X (gfx1250, wave32).
//   Phase 1: segmented sum/max/count reduction (register accum + LDS + atomics)
//   Phase 2: tiny MLP (64->32 relu ->64) on avg & max pooled [16,64] matrices
//            using v_wmma_f32_16x16x32_f16 in a single wave32, then sigmoid.
//   Phase 3: out = x * (1 + att[batch]) streamed as float4.
// HBM-bound: ~1.55 GB traffic -> ~67us floor at 23.3 TB/s.
// ---------------------------------------------------------------------------

#define NCH 64
#define NB  16
#define NH  32

typedef __attribute__((ext_vector_type(16))) _Float16 v16h;
typedef __attribute__((ext_vector_type(8)))  float    v8f;

// Monotone order-preserving float<->uint encoding for atomicMax on floats.
__device__ __forceinline__ unsigned enc_f32(float f) {
    unsigned b = __float_as_uint(f);
    return (b & 0x80000000u) ? ~b : (b | 0x80000000u);
}
__device__ __forceinline__ float dec_f32(unsigned u) {
    unsigned b = (u & 0x80000000u) ? (u & 0x7FFFFFFFu) : ~u;
    return __uint_as_float(b);
}

// -------------------------- Phase 0: workspace init -------------------------
__global__ void ca_init_kernel(float* gsums, unsigned* gmaxs, float* gcnts) {
    int t = threadIdx.x;
    for (int i = t; i < NB * NCH; i += blockDim.x) {
        gsums[i] = 0.0f;
        gmaxs[i] = 0u;            // enc-domain minimum (below enc(-inf))
    }
    if (t < NB) gcnts[t] = 0.0f;
}

// -------------------------- Phase 1: segmented reduce -----------------------
// Block of 256 threads owns a contiguous row chunk. Thread t handles channel
// c = t&63 of rows (t>>6) + 4k. Sorted batch_ids => register accumulation with
// rare flushes into LDS (float add / uint max atomics), then one global
// atomic merge per touched (batch,channel).
__global__ void ca_reduce_kernel(const float* __restrict__ x,
                                 const int*   __restrict__ bid,
                                 float* gsums, unsigned* gmaxs, float* gcnts,
                                 int N) {
    __shared__ float    sSum[NB * NCH];
    __shared__ unsigned sMax[NB * NCH];
    __shared__ float    sCnt[NB];

    for (int i = threadIdx.x; i < NB * NCH; i += blockDim.x) {
        sSum[i] = 0.0f;
        sMax[i] = 0u;
    }
    if (threadIdx.x < NB) sCnt[threadIdx.x] = 0.0f;
    __syncthreads();

    long long chunk   = ((long long)N + gridDim.x - 1) / gridDim.x;
    long long r_begin = (long long)blockIdx.x * chunk;
    long long r_end   = r_begin + chunk;
    if (r_end > N) r_end = N;

    const int c  = threadIdx.x & 63;
    const int r0 = threadIdx.x >> 6;              // 0..3

    float sumv = 0.0f, maxv = -3.402823466e38f;
    int   curb = -1, cnt = 0;

    for (long long r = r_begin + r0; r < r_end; r += 4) {
        int   b = bid[r];
        float v = x[r * (long long)NCH + c];
        if (b != curb) {
            if (curb >= 0) {
                atomicAdd(&sSum[curb * NCH + c], sumv);
                atomicMax(&sMax[curb * NCH + c], enc_f32(maxv));
                if (c == 0) atomicAdd(&sCnt[curb], (float)cnt);
            }
            curb = b; sumv = 0.0f; maxv = -3.402823466e38f; cnt = 0;
        }
        sumv += v;
        maxv  = fmaxf(maxv, v);
        ++cnt;
    }
    if (curb >= 0) {
        atomicAdd(&sSum[curb * NCH + c], sumv);
        atomicMax(&sMax[curb * NCH + c], enc_f32(maxv));
        if (c == 0) atomicAdd(&sCnt[curb], (float)cnt);
    }
    __syncthreads();

    for (int i = threadIdx.x; i < NB * NCH; i += blockDim.x) {
        int b = i >> 6;
        if (sCnt[b] > 0.0f) {
            atomicAdd(&gsums[i], sSum[i]);
            atomicMax(&gmaxs[i], sMax[i]);
        }
    }
    if (threadIdx.x < NB && sCnt[threadIdx.x] > 0.0f)
        atomicAdd(&gcnts[threadIdx.x], sCnt[threadIdx.x]);
}

// -------------------------- Phase 2: WMMA MLP -------------------------------
// A operand: 16x32 f16 tile, per ISA 7.12.2 layout (lane M, half-select,
// 8 VGPRs x 2 f16 K-values). B operand: mirrored layout (per-lane column).
__device__ __forceinline__ v16h load_a_tile(const float* src, int ld, int k0,
                                            int lane) {
    int M = lane & 15, half = lane >> 4;
    v16h a;
#pragma unroll
    for (int v = 0; v < 8; ++v) {
        int grp = v >> 2, w = v & 3;
        int k = k0 + grp * 16 + half * 8 + w * 2;
        a[2 * v]     = (_Float16)src[M * ld + k];
        a[2 * v + 1] = (_Float16)src[M * ld + k + 1];
    }
    return a;
}

// B element (k, n) = W[(n_base + n) * ld + k_base + k]  (W row-major, B = W^T)
__device__ __forceinline__ v16h load_b_tile(const float* __restrict__ W, int ld,
                                            int n_base, int k_base, int lane) {
    int n = lane & 15, half = lane >> 4;
    const float* row = W + (size_t)(n_base + n) * ld + k_base;
    v16h b;
#pragma unroll
    for (int v = 0; v < 8; ++v) {
        int grp = v >> 2, w = v & 3;
        int k = grp * 16 + half * 8 + w * 2;
        b[2 * v]     = (_Float16)row[k];
        b[2 * v + 1] = (_Float16)row[k + 1];
    }
    return b;
}

__global__ void ca_mlp_kernel(const float* __restrict__ W1,
                              const float* __restrict__ b1,
                              const float* __restrict__ W2,
                              const float* __restrict__ b2,
                              const float* __restrict__ gsums,
                              const unsigned* __restrict__ gmaxs,
                              const float* __restrict__ gcnts,
                              float* __restrict__ att_out) {
    __shared__ float sIn[2][NB * NCH];   // [0]=avg, [1]=max
    __shared__ float sH[16 * NH];        // layer-1 activations
    __shared__ float sAcc[NB * NCH];     // mlp(avg)+mlp(max)

    const int lane = threadIdx.x;        // single wave32
    for (int i = lane; i < NB * NCH; i += 32) {
        int   b   = i >> 6;
        float cnt = fmaxf(gcnts[b], 1.0f);
        sIn[0][i] = gsums[i] / cnt;
        sIn[1][i] = dec_f32(gmaxs[i]);
        sAcc[i]   = 0.0f;
    }
    __syncthreads();

    const int Ncol = lane & 15, half = lane >> 4;

    for (int pass = 0; pass < 2; ++pass) {
        const float* P = sIn[pass];

        // Layer 1: [16,64] @ W1^T[64,32] + b1, ReLU  -> sH [16,32]
#pragma unroll
        for (int nj = 0; nj < 2; ++nj) {
            v8f acc;
            float bias = b1[nj * 16 + Ncol];
#pragma unroll
            for (int v = 0; v < 8; ++v) acc[v] = bias;
#pragma unroll
            for (int ki = 0; ki < 2; ++ki) {
                v16h a  = load_a_tile(P, NCH, ki * 32, lane);
                v16h bm = load_b_tile(W1, NCH, nj * 16, ki * 32, lane);
                acc = __builtin_amdgcn_wmma_f32_16x16x32_f16(
                    false, a, false, bm, (short)0, acc, false, false);
            }
#pragma unroll
            for (int v = 0; v < 8; ++v) {
                int row = v + 8 * half;              // C/D layout: M = v(+8)
                sH[row * NH + nj * 16 + Ncol] = fmaxf(acc[v], 0.0f);
            }
        }
        __syncthreads();

        // Layer 2: [16,32] @ W2^T[32,64] + b2 -> accumulate into sAcc
#pragma unroll
        for (int nj = 0; nj < 4; ++nj) {
            v8f acc;
            float bias = b2[nj * 16 + Ncol];
#pragma unroll
            for (int v = 0; v < 8; ++v) acc[v] = bias;
            v16h a  = load_a_tile(sH, NH, 0, lane);
            v16h bm = load_b_tile(W2, NH, nj * 16, 0, lane);
            acc = __builtin_amdgcn_wmma_f32_16x16x32_f16(
                false, a, false, bm, (short)0, acc, false, false);
#pragma unroll
            for (int v = 0; v < 8; ++v) {
                int row = v + 8 * half;
                sAcc[row * NCH + nj * 16 + Ncol] += acc[v];
            }
        }
        __syncthreads();
    }

    for (int i = lane; i < NB * NCH; i += 32)
        att_out[i] = 1.0f / (1.0f + __expf(-sAcc[i]));
}

// -------------------------- Phase 3: apply ----------------------------------
// out = x * (1 + att[batch]); streamed float4 (16 float4 per row).
__global__ void ca_apply_kernel(const float* __restrict__ x,
                                const int*   __restrict__ bid,
                                const float* __restrict__ att,
                                float* __restrict__ out,
                                long long total4) {
    __shared__ float sAtt[NB * NCH];
    for (int i = threadIdx.x; i < NB * NCH; i += blockDim.x)
        sAtt[i] = 1.0f + att[i];
    __syncthreads();

    long long stride = (long long)gridDim.x * blockDim.x;
    for (long long idx = (long long)blockIdx.x * blockDim.x + threadIdx.x;
         idx < total4; idx += stride) {
        long long row = idx >> 4;
        int       c4  = (int)(idx & 15);
        int       b   = bid[row];
        const float* s = &sAtt[b * NCH + c4 * 4];
        float4 v = ((const float4*)x)[idx];
        v.x *= s[0];
        v.y *= s[1];
        v.z *= s[2];
        v.w *= s[3];
        ((float4*)out)[idx] = v;
    }
}

// ---------------------------------------------------------------------------
extern "C" void kernel_launch(void* const* d_in, const int* in_sizes, int n_in,
                              void* d_out, int out_size, void* d_ws, size_t ws_size,
                              hipStream_t stream) {
    const float* x   = (const float*)d_in[0];   // [N, 64]
    const int*   bid = (const int*)d_in[1];     // [N]
    const float* W1  = (const float*)d_in[2];   // [32, 64]
    const float* b1  = (const float*)d_in[3];   // [32]
    const float* W2  = (const float*)d_in[4];   // [64, 32]
    const float* b2  = (const float*)d_in[5];   // [64]
    float*       out = (float*)d_out;

    const int N = in_sizes[1];

    // Workspace layout (needs ~12.5 KB)
    float*    gsums = (float*)d_ws;                      // 1024 f32
    unsigned* gmaxs = (unsigned*)(gsums + NB * NCH);     // 1024 u32
    float*    gcnts = (float*)(gmaxs + NB * NCH);        // 16 f32
    float*    gatt  = gcnts + NB;                        // 1024 f32

    ca_init_kernel<<<1, 256, 0, stream>>>(gsums, gmaxs, gcnts);

    ca_reduce_kernel<<<1024, 256, 0, stream>>>(x, bid, gsums, gmaxs, gcnts, N);

    ca_mlp_kernel<<<1, 32, 0, stream>>>(W1, b1, W2, b2, gsums, gmaxs, gcnts, gatt);

    long long total4 = (long long)N * (NCH / 4);
    ca_apply_kernel<<<2048, 256, 0, stream>>>(x, bid, gatt, out, total4);
}
